// Transformer4Gen_4973572129402
// MI455X (gfx1250) — compile-verified, matrix operand order checked by reading, not run on hardware
//
#include <hip/hip_runtime.h>

// ---------------------------------------------------------------------------
// CDNA5 (gfx1250) wave32 WMMA transformer forward.
// All matmuls via v_wmma_f32_16x16x32_bf16 (f32 accumulate).
// Softmax cross-lane reductions via v_permlane16_b32 (VALU, co-executes with
// WMMA) instead of ds_bpermute, keeping the LDS pipe free for the P-bounce.
// ---------------------------------------------------------------------------

typedef __attribute__((ext_vector_type(16))) __bf16 v16bf;
typedef __attribute__((ext_vector_type(8)))  float  v8f;
typedef unsigned short u16;

__device__ __forceinline__ u16 f2bf(float f) {
  unsigned u = __float_as_uint(f);
  u += 0x7FFFu + ((u >> 16) & 1u);          // round-to-nearest-even
  return (u16)(u >> 16);
}

__device__ __forceinline__ v16bf ldfrag(const u16* p) {
  return *reinterpret_cast<const v16bf*>(p); // 32B aligned by construction
}

__device__ __forceinline__ v8f wmma_bf16(v16bf a, v16bf b, v8f c) {
  // (neg_a, A, neg_b, B, c_mod, C, reuse_a, reuse_b)
  return __builtin_amdgcn_wmma_f32_16x16x32_bf16(false, a, false, b,
                                                 (short)0, c, false, false);
}

__device__ __forceinline__ v8f v8zero() {
  v8f z = {0.f,0.f,0.f,0.f,0.f,0.f,0.f,0.f};
  return z;
}

// xor-shuffle within each 16-lane row, on the VALU (v_permlane16_b32).
// sel nibble j = j^m; sel0 = entries 0..7, sel1 = entries 8..15.
template<unsigned S0, unsigned S1, int M>
__device__ __forceinline__ float lane_xor16(float v) {
#if __has_builtin(__builtin_amdgcn_permlane16)
  return __int_as_float(__builtin_amdgcn_permlane16(
      __float_as_int(v), __float_as_int(v), (int)S0, (int)S1, false, false));
#else
  return __shfl_xor(v, M, 16);
#endif
}
__device__ __forceinline__ float redmax16(float v) {
  v = fmaxf(v, lane_xor16<0x67452301u, 0xEFCDAB89u, 1>(v));
  v = fmaxf(v, lane_xor16<0x54761032u, 0xDCFE98BAu, 2>(v));
  v = fmaxf(v, lane_xor16<0x32107654u, 0xBA98FEDCu, 4>(v));
  v = fmaxf(v, lane_xor16<0xFEDCBA98u, 0x76543210u, 8>(v));
  return v;
}
__device__ __forceinline__ float redsum16(float v) {
  v += lane_xor16<0x67452301u, 0xEFCDAB89u, 1>(v);
  v += lane_xor16<0x54761032u, 0xDCFE98BAu, 2>(v);
  v += lane_xor16<0x32107654u, 0xBA98FEDCu, 4>(v);
  v += lane_xor16<0xFEDCBA98u, 0x76543210u, 8>(v);
  return v;
}

// ---------------------------------------------------------------------------
// GEMM: C[M,N] = act(A_bf16[M,K] * BT_bf16[N,K]^T + bias (+ resid))
// Block = 256 threads (8 waves). Workgroup tile: 128(M) x 128(N);
// each wave: 16x128 (8 WMMA accumulators), K stepped by 32.
// C/D layout (ISA 7.12.2): vgpr i, lane l -> row = i + 8*(l>>4), col = l&15.
// ---------------------------------------------------------------------------
template<int RELU, int OUTBF, int RESID>
__global__ __launch_bounds__(256) void gemm_wmma_k(
    const u16* __restrict__ A, const u16* __restrict__ BT,
    const float* __restrict__ bias, const float* __restrict__ resid,
    float* __restrict__ Cf, u16* __restrict__ Cb, int K, int N) {
  const int lane = threadIdx.x & 31, wave = threadIdx.x >> 5;
  const int lr = lane & 15, lh = lane >> 4;
  const int m0 = blockIdx.x * 128 + wave * 16;
  const int n0 = blockIdx.y * 128;

  v8f acc[8];
#pragma unroll
  for (int t = 0; t < 8; ++t) acc[t] = v8zero();

  const u16* Ap = A  + (size_t)(m0 + lr) * K + lh * 16;
  const u16* Bp = BT + (size_t)(n0 + lr) * K + lh * 16;

  for (int k0 = 0; k0 < K; k0 += 32) {
    v16bf a = ldfrag(Ap + k0);
    __builtin_prefetch(Ap + k0 + 128, 0, 0);   // global_prefetch_b8
#pragma unroll
    for (int t = 0; t < 8; ++t) {
      v16bf b = ldfrag(Bp + (size_t)(t * 16) * K + k0);
      acc[t] = wmma_bf16(a, b, acc[t]);
    }
  }

#pragma unroll
  for (int t = 0; t < 8; ++t) {
    const int col = n0 + t * 16 + lr;
    const float bs = bias[col];
#pragma unroll
    for (int i = 0; i < 8; ++i) {
      const int row = m0 + i + 8 * lh;
      float v = acc[t][i] + bs;
      if (RESID) v += resid[(size_t)row * N + col];
      if (RELU)  v = fmaxf(v, 0.0f);
      if (OUTBF) Cb[(size_t)row * N + col] = f2bf(v);
      else       Cf[(size_t)row * N + col] = v;
    }
  }
}

// ---------------------------------------------------------------------------
// Fused attention (flash-style, online softmax), one wave = 16 query rows.
// Q,K: [B*S, 512] bf16, head h in cols [h*64, h*64+64). VT: [B,H,64,S] bf16.
// O[b,s,h*64+d] = softmax(Q K^T / 8) V  (all 1024 keys streamed, 32 at a time)
// ---------------------------------------------------------------------------
__global__ __launch_bounds__(256) void attn_wmma_k(
    const u16* __restrict__ Q, const u16* __restrict__ Krow,
    const u16* __restrict__ VT, u16* __restrict__ O, float scale) {
  constexpr int S = 1024, E = 512, D = 64;
  __shared__ __align__(32) u16 pbuf[8][16 * 32];  // per-wave P-tile bounce

  const int lane = threadIdx.x & 31, wave = threadIdx.x >> 5;
  const int lr = lane & 15, lh = lane >> 4;
  const int bh = blockIdx.y, b = bh >> 3, h = bh & 7;
  const int q0 = (blockIdx.x * 8 + wave) * 16;

  const u16* Qb = Q    + (size_t)b * S * E + h * D;
  const u16* Kb = Krow + (size_t)b * S * E + h * D;
  const u16* Vt = VT   + (size_t)bh * D * S;
  u16* pw = pbuf[wave];

  // Q fragments (A role, 16x64 split as two 16x32 frags)
  v16bf qa0 = ldfrag(Qb + (size_t)(q0 + lr) * E +      lh * 16);
  v16bf qa1 = ldfrag(Qb + (size_t)(q0 + lr) * E + 32 + lh * 16);

  v8f o0 = v8zero(), o1 = v8zero(), o2 = v8zero(), o3 = v8zero();
  float rm[8], rl[8];
#pragma unroll
  for (int i = 0; i < 8; ++i) { rm[i] = -3.0e38f; rl[i] = 0.f; }

  for (int j0 = 0; j0 < S; j0 += 32) {
    // K^T fragments (B role): key cols j0..j0+31, contraction over 64 dims
    v16bf k00 = ldfrag(Kb + (size_t)(j0      + lr) * E +      lh * 16);
    v16bf k01 = ldfrag(Kb + (size_t)(j0      + lr) * E + 32 + lh * 16);
    v16bf k10 = ldfrag(Kb + (size_t)(j0 + 16 + lr) * E +      lh * 16);
    v16bf k11 = ldfrag(Kb + (size_t)(j0 + 16 + lr) * E + 32 + lh * 16);

    v8f s0 = v8zero(); s0 = wmma_bf16(qa0, k00, s0); s0 = wmma_bf16(qa1, k01, s0);
    v8f s1 = v8zero(); s1 = wmma_bf16(qa0, k10, s1); s1 = wmma_bf16(qa1, k11, s1);

    float alpha[8];
#pragma unroll
    for (int i = 0; i < 8; ++i) {
      float a0 = s0[i] * scale, a1 = s1[i] * scale;
      const float tmax = redmax16(fmaxf(a0, a1));   // row lives across 16 lanes
      const float mn = fmaxf(rm[i], tmax);
      const float p0 = __expf(a0 - mn), p1 = __expf(a1 - mn);
      const float al = __expf(rm[i] - mn);
      const float rs = redsum16(p0 + p1);
      rl[i] = rl[i] * al + rs; rm[i] = mn; alpha[i] = al;
      s0[i] = p0; s1[i] = p1;
    }
#pragma unroll
    for (int i = 0; i < 8; ++i) {
      o0[i] *= alpha[i]; o1[i] *= alpha[i];
      o2[i] *= alpha[i]; o3[i] *= alpha[i];
    }

    // C-layout -> A-fragment relayout of P (16x32) through per-wave LDS
#pragma unroll
    for (int i = 0; i < 8; ++i) {
      pw[(i + 8 * lh) * 32 +      lr] = f2bf(s0[i]);
      pw[(i + 8 * lh) * 32 + 16 + lr] = f2bf(s1[i]);
    }
    asm volatile("s_wait_dscnt 0" ::: "memory");
    v16bf pa = *reinterpret_cast<const v16bf*>(pw + lr * 32 + lh * 16);

    // V fragments (B role) from VT[d][s]: K-contiguous in key index
    v16bf vb0 = ldfrag(Vt + (size_t)( 0 + lr) * S + j0 + lh * 16);
    v16bf vb1 = ldfrag(Vt + (size_t)(16 + lr) * S + j0 + lh * 16);
    v16bf vb2 = ldfrag(Vt + (size_t)(32 + lr) * S + j0 + lh * 16);
    v16bf vb3 = ldfrag(Vt + (size_t)(48 + lr) * S + j0 + lh * 16);
    o0 = wmma_bf16(pa, vb0, o0); o1 = wmma_bf16(pa, vb1, o1);
    o2 = wmma_bf16(pa, vb2, o2); o3 = wmma_bf16(pa, vb3, o3);
  }

#pragma unroll
  for (int i = 0; i < 8; ++i) {
    const float inv = 1.0f / rl[i];
    const int row = b * S + q0 + i + 8 * lh;
    const size_t base = (size_t)row * E + h * D + lr;
    O[base +  0] = f2bf(o0[i] * inv);
    O[base + 16] = f2bf(o1[i] * inv);
    O[base + 32] = f2bf(o2[i] * inv);
    O[base + 48] = f2bf(o3[i] * inv);
  }
}

// ---------------------------------------------------------------------------
// LayerNorm over last dim (512), one wave32 per row; writes f32 + bf16 mirror
// ---------------------------------------------------------------------------
__global__ __launch_bounds__(256) void layernorm_k(
    const float* __restrict__ X, const float* __restrict__ g,
    const float* __restrict__ bta, float* __restrict__ Y, u16* __restrict__ Yb) {
  constexpr int E = 512;
  const int lane = threadIdx.x & 31, wave = threadIdx.x >> 5;
  const int row = blockIdx.x * 8 + wave;
  const float* x = X + (size_t)row * E;

  float v[16]; float s = 0.f;
#pragma unroll
  for (int i = 0; i < 16; ++i) { v[i] = x[i * 32 + lane]; s += v[i]; }
#pragma unroll
  for (int m = 1; m < 32; m <<= 1) s += __shfl_xor(s, m, 32);
  const float mean = s * (1.0f / E);

  float vs = 0.f;
#pragma unroll
  for (int i = 0; i < 16; ++i) { const float d = v[i] - mean; vs += d * d; }
#pragma unroll
  for (int m = 1; m < 32; m <<= 1) vs += __shfl_xor(vs, m, 32);
  const float rstd = rsqrtf(vs * (1.0f / E) + 1e-5f);

#pragma unroll
  for (int i = 0; i < 16; ++i) {
    const int c = i * 32 + lane;
    const float o = (v[i] - mean) * rstd * g[c] + bta[c];
    Y [(size_t)row * E + c] = o;
    Yb[(size_t)row * E + c] = f2bf(o);
  }
}

// ---------------------------------------------------------------------------
// Elementwise / repack kernels
// ---------------------------------------------------------------------------
__global__ void copy_cvt_k(const float* __restrict__ X,
                           float* __restrict__ Yf, u16* __restrict__ Yb) {
  const int i = blockIdx.x * 256 + threadIdx.x;
  const float v = X[i];
  Yf[i] = v; Yb[i] = f2bf(v);
}

// W [8,512,64] f32 -> WT [512(n=h*64+d), 512(e)] bf16
__global__ void repack_head_w_k(const float* __restrict__ W, u16* __restrict__ WT) {
  const int idx = blockIdx.x * 256 + threadIdx.x;  // n*512 + e
  const int n = idx >> 9, e = idx & 511;
  const int h = n >> 6, d = n & 63;
  WT[idx] = f2bf(W[((size_t)(h * 512 + e)) * 64 + d]);
}

// W [Kd,Nd] f32 -> WT [Nd,Kd] bf16
__global__ void transpose_w_k(const float* __restrict__ W, u16* __restrict__ WT,
                              int Kd, int Nd) {
  const int idx = blockIdx.x * 256 + threadIdx.x;  // n*Kd + k
  const int n = idx / Kd, k = idx % Kd;
  WT[idx] = f2bf(W[(size_t)k * Nd + n]);
}

// V [4096,512] bf16 -> VT [B,H,64,S] bf16 (coalesced in s)
__global__ void transpose_v_k(const u16* __restrict__ V, u16* __restrict__ VT) {
  const int idx = blockIdx.x * 256 + threadIdx.x;  // ((bh*64)+d)*1024 + s
  const int s = idx & 1023;
  const int r = idx >> 10;
  const int d = r & 63;
  const int bh = r >> 6;
  const int b = bh >> 3, h = bh & 7;
  VT[idx] = V[((size_t)(b * 1024 + s)) * 512 + h * 64 + d];
}

// ---------------------------------------------------------------------------
// Host orchestration
// ---------------------------------------------------------------------------
extern "C" void kernel_launch(void* const* d_in, const int* in_sizes, int n_in,
                              void* d_out, int out_size, void* d_ws, size_t ws_size,
                              hipStream_t stream) {
  (void)in_sizes; (void)n_in; (void)out_size; (void)ws_size;
  constexpr int Bb = 4, S = 1024, E = 512, H = 8, D = 64, F = 256;
  constexpr int M = Bb * S;  // 4096 token rows

  auto F32 = [&](int i) { return (const float*)d_in[i]; };

  char* base = (char*)d_ws; size_t off = 0;
  auto alloc = [&](size_t bytes) -> void* {
    off = (off + 255) & ~(size_t)255;
    void* p = base + off; off += bytes; return p;
  };

  float* x_f   = (float*)alloc((size_t)M * E * 4);
  float* tmp_f = (float*)alloc((size_t)M * E * 4);
  u16* x_b   = (u16*)alloc((size_t)M * E * 2);
  u16* enc_b = (u16*)alloc((size_t)M * E * 2);
  u16* q_b   = (u16*)alloc((size_t)M * E * 2);
  u16* k_b   = (u16*)alloc((size_t)M * E * 2);
  u16* v_b   = (u16*)alloc((size_t)M * E * 2);
  u16* vt_b  = (u16*)alloc((size_t)M * E * 2);
  u16* o_b   = (u16*)alloc((size_t)M * E * 2);
  u16* h1_b  = (u16*)alloc((size_t)M * F * 2);

  struct MhaW { u16 *qT, *kT, *vT, *oT; const float *bq, *bk, *bv, *bo; };
  struct FfW  { u16 *w1T, *w2T; const float *b1, *b2; };

  auto repack_mha = [&](int i0) {
    MhaW w;
    w.qT = (u16*)alloc((size_t)E * E * 2); w.kT = (u16*)alloc((size_t)E * E * 2);
    w.vT = (u16*)alloc((size_t)E * E * 2); w.oT = (u16*)alloc((size_t)E * E * 2);
    repack_head_w_k<<<(E * E) / 256, 256, 0, stream>>>(F32(i0 + 0), w.qT);
    repack_head_w_k<<<(E * E) / 256, 256, 0, stream>>>(F32(i0 + 2), w.kT);
    repack_head_w_k<<<(E * E) / 256, 256, 0, stream>>>(F32(i0 + 4), w.vT);
    transpose_w_k  <<<(E * E) / 256, 256, 0, stream>>>(F32(i0 + 6), w.oT, E, E);
    w.bq = F32(i0 + 1); w.bk = F32(i0 + 3); w.bv = F32(i0 + 5); w.bo = F32(i0 + 7);
    return w;
  };
  auto repack_ff = [&](int i0) {
    FfW w;
    w.w1T = (u16*)alloc((size_t)E * F * 2);
    w.w2T = (u16*)alloc((size_t)F * E * 2);
    transpose_w_k<<<(E * F) / 256, 256, 0, stream>>>(F32(i0 + 0), w.w1T, E, F);
    transpose_w_k<<<(F * E) / 256, 256, 0, stream>>>(F32(i0 + 2), w.w2T, F, E);
    w.b1 = F32(i0 + 1); w.b2 = F32(i0 + 3);
    return w;
  };

  // GEMM launchers (grid: M/128 x N/128, block 256 = 8 waves)
  auto gemm_bf = [&](const u16* A, const u16* BT, const float* bias, u16* Cb,
                     int K, int N, int relu) {
    dim3 grid(M / 128, N / 128), blk(256);
    if (relu) gemm_wmma_k<1, 1, 0><<<grid, blk, 0, stream>>>(A, BT, bias, nullptr, nullptr, Cb, K, N);
    else      gemm_wmma_k<0, 1, 0><<<grid, blk, 0, stream>>>(A, BT, bias, nullptr, nullptr, Cb, K, N);
  };
  auto gemm_res = [&](const u16* A, const u16* BT, const float* bias,
                      const float* resid, float* Cf, int K, int N) {
    dim3 grid(M / 128, N / 128), blk(256);
    gemm_wmma_k<0, 0, 1><<<grid, blk, 0, stream>>>(A, BT, bias, resid, Cf, nullptr, K, N);
  };
  auto ln = [&](const float* X, const float* g, const float* bt, float* Yf, u16* Yb) {
    layernorm_k<<<M / 8, 256, 0, stream>>>(X, g, bt, Yf, Yb);
  };
  auto attn = [&](const u16* Qr, const u16* Kr, const u16* Vr, u16* Ob) {
    transpose_v_k<<<(Bb * H * D * S) / 256, 256, 0, stream>>>(Vr, vt_b);
    attn_wmma_k<<<dim3(S / 128, Bb * H), 256, 0, stream>>>(Qr, Kr, vt_b, Ob, 0.125f);
  };

  // ---- repack all weights (bf16 + transposed, K-contiguous operands) ----
  // input order: seq, tgt, 2x enc layer (16 arrays), 2x dec layer (26 arrays)
  MhaW eAt[2] = { repack_mha(2 + 0 * 16), repack_mha(2 + 1 * 16) };
  FfW  eFf[2] = { repack_ff (2 + 0 * 16 + 10), repack_ff (2 + 1 * 16 + 10) };
  const int db = 2 + 2 * 16;  // 34
  MhaW dSa[2] = { repack_mha(db + 0 * 26),      repack_mha(db + 1 * 26) };
  MhaW dIa[2] = { repack_mha(db + 0 * 26 + 10), repack_mha(db + 1 * 26 + 10) };
  FfW  dFf[2] = { repack_ff (db + 0 * 26 + 20), repack_ff (db + 1 * 26 + 20) };

  // -------------------------------- encoder --------------------------------
  copy_cvt_k<<<(M * E) / 256, 256, 0, stream>>>(F32(0), x_f, x_b);
  for (int L = 0; L < 2; ++L) {
    const int p = 2 + L * 16;
    const MhaW& w = eAt[L];
    gemm_bf(x_b, w.qT, w.bq, q_b, E, E, 0);
    gemm_bf(x_b, w.kT, w.bk, k_b, E, E, 0);
    gemm_bf(x_b, w.vT, w.bv, v_b, E, E, 0);
    attn(q_b, k_b, v_b, o_b);
    gemm_res(o_b, w.oT, w.bo, x_f, tmp_f, E, E);
    ln(tmp_f, F32(p + 8), F32(p + 9), x_f, x_b);

    gemm_bf(x_b, eFf[L].w1T, eFf[L].b1, h1_b, E, F, 1);         // ReLU
    gemm_res(h1_b, eFf[L].w2T, eFf[L].b2, x_f, tmp_f, F, E);
    u16* yb = (L == 1) ? enc_b : x_b;  // final encoder bf16 -> enc_b
    ln(tmp_f, F32(p + 14), F32(p + 15), x_f, yb);
  }

  // -------------------------------- decoder --------------------------------
  copy_cvt_k<<<(M * E) / 256, 256, 0, stream>>>(F32(1), x_f, x_b);
  for (int L = 0; L < 2; ++L) {
    const int p = db + L * 26;
    {  // self-attention
      const MhaW& w = dSa[L];
      gemm_bf(x_b, w.qT, w.bq, q_b, E, E, 0);
      gemm_bf(x_b, w.kT, w.bk, k_b, E, E, 0);
      gemm_bf(x_b, w.vT, w.bv, v_b, E, E, 0);
      attn(q_b, k_b, v_b, o_b);
      gemm_res(o_b, w.oT, w.bo, x_f, tmp_f, E, E);
      ln(tmp_f, F32(p + 8), F32(p + 9), x_f, x_b);
    }
    {  // cross-attention: Q,V from enc; K from x; softmax over enc positions
      const MhaW& w = dIa[L];
      gemm_bf(enc_b, w.qT, w.bq, q_b, E, E, 0);
      gemm_bf(x_b,   w.kT, w.bk, k_b, E, E, 0);
      gemm_bf(enc_b, w.vT, w.bv, v_b, E, E, 0);
      attn(k_b, q_b, v_b, o_b);   // queries=K(x), keys=Q(enc), values=V(enc)
      gemm_res(o_b, w.oT, w.bo, x_f, tmp_f, E, E);
      ln(tmp_f, F32(p + 18), F32(p + 19), x_f, x_b);
    }
    gemm_bf(x_b, dFf[L].w1T, dFf[L].b1, h1_b, E, F, 1);
    gemm_res(h1_b, dFf[L].w2T, dFf[L].b2, x_f, tmp_f, F, E);
    float* Yf = (L == 1) ? (float*)d_out : x_f;   // final LN -> d_out (f32)
    ln(tmp_f, F32(p + 24), F32(p + 25), Yf, x_b);
  }
}
